// MSDeformAttn_18021682774368
// MI455X (gfx1250) — matrix operand order repaired and checked by reading, not compiled, not run on hardware
//
#include <hip/hip_runtime.h>
#include <hip/hip_bf16.h>

typedef __attribute__((ext_vector_type(16))) __bf16 v16bf;
typedef __attribute__((ext_vector_type(8)))  __bf16 v8bf;
typedef __attribute__((ext_vector_type(8)))  float  v8f;
typedef unsigned int uint32x4 __attribute__((ext_vector_type(4)));
typedef int          int32x8  __attribute__((ext_vector_type(8)));
typedef int          int32x4  __attribute__((ext_vector_type(4)));

#define K_DIM 256

#if __has_builtin(__builtin_amdgcn_tensor_load_to_lds)
#define USE_TDM 1
#else
#define USE_TDM 0
#endif

// ---------------------------------------------------------------------------
// Transpose + fp32->bf16 convert weight [K_DIM x C] -> Wt [C x K_DIM]
// so WMMA B fragments are contiguous 32B runs along K.
// ---------------------------------------------------------------------------
__global__ void prep_weight_kernel(const float* __restrict__ W,
                                   __bf16* __restrict__ Wt, int C) {
    int idx = blockIdx.x * blockDim.x + threadIdx.x;
    if (idx >= K_DIM * C) return;
    int c = idx / K_DIM;
    int k = idx - c * K_DIM;
    Wt[(size_t)c * K_DIM + k] = (__bf16)W[(size_t)k * C + c];
}

// ---------------------------------------------------------------------------
// fp32 -> bf16 activation convert, 8 elements/thread (16B stores).
// ---------------------------------------------------------------------------
__global__ void cvt_act_kernel(const float* __restrict__ in,
                               __bf16* __restrict__ out, int n8) {
    int idx = blockIdx.x * blockDim.x + threadIdx.x;
    if (idx >= n8) return;
    const float4* p = (const float4*)(in + (size_t)idx * 8);
    float4 a = p[0], b = p[1];
    v8bf o;
    o[0] = (__bf16)a.x; o[1] = (__bf16)a.y; o[2] = (__bf16)a.z; o[3] = (__bf16)a.w;
    o[4] = (__bf16)b.x; o[5] = (__bf16)b.y; o[6] = (__bf16)b.z; o[7] = (__bf16)b.w;
    *(v8bf*)(out + (size_t)idx * 8) = o;
}

// ---------------------------------------------------------------------------
// C[rows x cols] = A[rows x 256](bf16) @ Wt[cols x 256](bf16) + bias
// Block = 8 waves = 128 rows x 64 cols. The 32KB B tile for the block is
// DMA'd into LDS by ONE tensor_load_to_lds (TDM) descriptor, then every
// wave runs 32x v_wmma_f32_16x16x32_bf16 with B fragments from LDS.
// ---------------------------------------------------------------------------
__global__ __launch_bounds__(256) void gemm_wmma_kernel(
    const __bf16* __restrict__ A, const __bf16* __restrict__ Wt,
    const float* __restrict__ bias, float* __restrict__ C,
    int rows, int cols) {
    __shared__ __bf16 sB[64 * K_DIM];       // 32 KB of 320 KB/WGP
    const int lane = threadIdx.x & 31;
    const int wave = threadIdx.x >> 5;
    const int half = lane >> 4;   // 0: lanes 0-15, 1: lanes 16-31
    const int l15  = lane & 15;
    const int colBase = blockIdx.x * 64;

#if USE_TDM
    if (threadIdx.x < 32) {       // one TDM issue per block (EXEC ignored)
        uint64_t gaddr = (uint64_t)(uintptr_t)(Wt + (size_t)colBase * K_DIM);
        uint32_t ldsa  = (uint32_t)(uintptr_t)(&sB[0]);   // addr[31:0] = LDS offset
        // D# group0: count=1 | lds_addr | global_addr[56:0] | type=2
        uint32x4 g0 = { 1u, ldsa,
                        (unsigned)(gaddr & 0xffffffffu),
                        (unsigned)((gaddr >> 32) & 0x01ffffffu) | 0x80000000u };
        // D# group1: data_size=2B; tensor_dim0=256, tensor_dim1=64;
        // tile_dim0=256, tile_dim1=64; tensor_dim0_stride=256
        int32x8 g1 = { 0x00010000, 0x01000000, 0x00400000, 0x01000000,
                       64, 256, 0, 0 };
        int32x4 gz = { 0, 0, 0, 0 };        // 2-D tensor: groups 2/3 unused
#if defined(__clang_major__) && (__clang_major__ >= 23)
        int32x8 gz8 = { 0, 0, 0, 0, 0, 0, 0, 0 };
        __builtin_amdgcn_tensor_load_to_lds(g0, g1, gz, gz, gz8, 0);
#else
        __builtin_amdgcn_tensor_load_to_lds(g0, g1, gz, gz, 0);
#endif
        __builtin_amdgcn_s_wait_tensorcnt(0);
    }
    __syncthreads();
#else
    {   // fallback: cooperative global->LDS copy of the B tile
        const __bf16* gB = Wt + (size_t)colBase * K_DIM;
        for (int i = threadIdx.x; i < 64 * K_DIM / 8; i += 256)
            *(v8bf*)(sB + i * 8) = *(const v8bf*)(gB + (size_t)i * 8);
        __syncthreads();
    }
#endif

    const int tileRow = (blockIdx.y * 8 + wave) * 16;
    if (tileRow >= rows) return;            // wave-uniform: EXEC stays all-1s
    int arow = tileRow + l15;
    if (arow >= rows) arow = rows - 1;      // clamp; masked at store
    const __bf16* Ar = A + (size_t)arow * K_DIM;
    const __bf16* Bs = sB + (size_t)l15 * K_DIM + half * 16;

    v8f acc[4] = {};
#pragma unroll
    for (int kb = 0; kb < K_DIM; kb += 32) {
        // A operand, contiguous k order (same as B): K = kb+half*16+{0..15}
        v8bf a_lo = *(const v8bf*)(Ar + kb + half * 16);
        v8bf a_hi = *(const v8bf*)(Ar + kb + half * 16 + 8);
        v16bf av = __builtin_shufflevector(a_lo, a_hi,
            0, 1, 2, 3, 4, 5, 6, 7, 8, 9, 10, 11, 12, 13, 14, 15);
#pragma unroll
        for (int t = 0; t < 4; ++t) {
            // B operand from LDS: lane col, K = kb + half*16 + {0..15}
            v16bf bv = *(const v16bf*)(Bs + (size_t)t * 16 * K_DIM + kb);
            acc[t] = __builtin_amdgcn_wmma_f32_16x16x32_bf16(
                false, av, false, bv, (short)0, acc[t], false, false);
        }
    }
#pragma unroll
    for (int t = 0; t < 4; ++t) {
        const int col = colBase + t * 16 + l15;
        const float bc = bias[col];
#pragma unroll
        for (int r = 0; r < 8; ++r) {       // C layout: VGPR r -> M = r + half*8
            int orow = tileRow + r + half * 8;
            if (orow < rows) C[(size_t)orow * cols + col] = acc[t][r] + bc;
        }
    }
}

// ---------------------------------------------------------------------------
// Deformable sampling: one thread per (n, q, m). Softmax over 16 logits,
// 4 levels x 4 points x 4-tap bilinear (zero-padding), D=32 accumulator.
// value (40.8 MB) fits in the 192 MB L2 -> gathers hit L2, not HBM.
// Output written as bf16 (A operand of the final WMMA GEMM).
// ---------------------------------------------------------------------------
__global__ __launch_bounds__(256) void msda_sample_kernel(
    const float* __restrict__ value,   // [N, Len, 256]  (m*32+d)
    const float* __restrict__ offs,    // [N, Lq, 256]   (m*32 + l*8 + p*2 + xy)
    const float* __restrict__ attnl,   // [N, Lq, 128]   (m*16 + l*4 + p)
    const float* __restrict__ refp,    // [N, Lq, 4, 2]
    const int*   __restrict__ shapes,  // [4,2] (H, W)
    __bf16* __restrict__ outh,         // [N, Lq, 256] bf16
    int Lq, int Len, int total) {
    int gid = blockIdx.x * blockDim.x + threadIdx.x;
    if (gid >= total) return;
    int m  = gid & 7;
    int nq = gid >> 3;
    int n  = nq / Lq;

    const float* al = attnl + (size_t)nq * 128 + m * 16;
    float mx = -1e30f;
#pragma unroll
    for (int i = 0; i < 16; ++i) mx = fmaxf(mx, al[i]);
    float s = 0.f;
#pragma unroll
    for (int i = 0; i < 16; ++i) s += __expf(al[i] - mx);
    const float sinv = 1.f / s;

    const float* off = offs + (size_t)nq * 256 + m * 32;
    const float* ref = refp + (size_t)nq * 8;

    float acc[32];
#pragma unroll
    for (int d = 0; d < 32; ++d) acc[d] = 0.f;

    int start = 0;
    for (int l = 0; l < 4; ++l) {
        const int H = shapes[2 * l], W = shapes[2 * l + 1];
        const float fH = (float)H, fW = (float)W;
        const float rx = ref[2 * l], ry = ref[2 * l + 1];
        for (int p = 0; p < 4; ++p) {
            float lx = rx + off[l * 8 + p * 2 + 0] / fW;
            float ly = ry + off[l * 8 + p * 2 + 1] / fH;
            float x = lx * fW - 0.5f;
            float y = ly * fH - 0.5f;
            float xf = floorf(x), yf = floorf(y);
            int x0 = (int)xf, y0 = (int)yf;
            float wx = x - xf, wy = y - yf;
            float aw = __expf(al[l * 4 + p] - mx) * sinv;
            float tw[4] = { (1.f - wx) * (1.f - wy) * aw, wx * (1.f - wy) * aw,
                            (1.f - wx) * wy * aw,         wx * wy * aw };
            int xs[4] = { x0, x0 + 1, x0,     x0 + 1 };
            int ys[4] = { y0, y0,     y0 + 1, y0 + 1 };
#pragma unroll
            for (int t = 0; t < 4; ++t) {
                int xi = xs[t], yi = ys[t];
                if (xi < 0 || xi >= W || yi < 0 || yi >= H) continue;
                const float* vp = value +
                    ((size_t)(n * Len + start + yi * W + xi)) * 256 + m * 32;
                float tt = tw[t];
#pragma unroll
                for (int j = 0; j < 8; ++j) {       // 128B contiguous slab
                    float4 v = ((const float4*)vp)[j];
                    acc[4 * j + 0] += tt * v.x;
                    acc[4 * j + 1] += tt * v.y;
                    acc[4 * j + 2] += tt * v.z;
                    acc[4 * j + 3] += tt * v.w;
                }
            }
        }
        start += H * W;
    }
    __bf16* op = outh + (size_t)nq * 256 + m * 32;
#pragma unroll
    for (int j = 0; j < 4; ++j) {                   // 4 x 16B bf16 stores
        v8bf o;
#pragma unroll
        for (int i = 0; i < 8; ++i) o[i] = (__bf16)acc[8 * j + i];
        ((v8bf*)op)[j] = o;
    }
}

// ---------------------------------------------------------------------------
extern "C" void kernel_launch(void* const* d_in, const int* in_sizes, int n_in,
                              void* d_out, int out_size, void* d_ws, size_t ws_size,
                              hipStream_t stream) {
    const float* query = (const float*)d_in[0];
    const float* refp  = (const float*)d_in[1];
    const float* infl  = (const float*)d_in[2];
    const float* Wv = (const float*)d_in[3];
    const float* bv = (const float*)d_in[4];
    const float* Wo = (const float*)d_in[5];
    const float* bo = (const float*)d_in[6];
    const float* Wa = (const float*)d_in[7];
    const float* ba = (const float*)d_in[8];
    const float* Ww = (const float*)d_in[9];
    const float* bw = (const float*)d_in[10];
    const int* shapes = (const int*)d_in[11];

    const int N = 2;
    const int Lq  = in_sizes[0] / (N * 256);
    const int Len = in_sizes[2] / (N * 256);
    const int qrows = N * Lq;
    const int vrows = N * Len;

    char* ws = (char*)d_ws;
    __bf16* wtv = (__bf16*)(ws + 0);         // 256x256 bf16 = 128 KB
    __bf16* wto = (__bf16*)(ws + 131072);    // 256x256 bf16
    __bf16* wta = (__bf16*)(ws + 262144);    // 128x256 bf16 =  64 KB
    __bf16* wtw = (__bf16*)(ws + 327680);    // 256x256 bf16
    float*  value  = (float*)(ws + 458752);                 // [vrows, 256] fp32
    float*  offsb  = value + (size_t)vrows * 256;           // [qrows, 256] fp32
    float*  attnb  = offsb + (size_t)qrows * 256;           // [qrows, 128] fp32
    __bf16* outh   = (__bf16*)(attnb + (size_t)qrows * 128);// [qrows, 256] bf16
    __bf16* inflb  = outh  + (size_t)qrows * 256;           // [vrows, 256] bf16
    __bf16* queryb = inflb + (size_t)vrows * 256;           // [qrows, 256] bf16

    // weight transpose+convert, activation convert
    prep_weight_kernel<<<(65536 + 255) / 256, 256, 0, stream>>>(Wv, wtv, 256);
    prep_weight_kernel<<<(65536 + 255) / 256, 256, 0, stream>>>(Wo, wto, 256);
    prep_weight_kernel<<<(32768 + 255) / 256, 256, 0, stream>>>(Wa, wta, 128);
    prep_weight_kernel<<<(65536 + 255) / 256, 256, 0, stream>>>(Ww, wtw, 256);
    cvt_act_kernel<<<(vrows * 32 + 255) / 256, 256, 0, stream>>>(infl, inflb, vrows * 32);
    cvt_act_kernel<<<(qrows * 32 + 255) / 256, 256, 0, stream>>>(query, queryb, qrows * 32);

    auto rt8 = [](int rows) { return ((rows + 15) / 16 + 7) / 8; };

    gemm_wmma_kernel<<<dim3(4, rt8(vrows)), 256, 0, stream>>>(
        inflb, wtv, bv, value, vrows, 256);
    gemm_wmma_kernel<<<dim3(4, rt8(qrows)), 256, 0, stream>>>(
        queryb, wto, bo, offsb, qrows, 256);
    gemm_wmma_kernel<<<dim3(2, rt8(qrows)), 256, 0, stream>>>(
        queryb, wta, ba, attnb, qrows, 128);

    int total = qrows * 8;
    msda_sample_kernel<<<(total + 255) / 256, 256, 0, stream>>>(
        value, offsb, attnb, refp, shapes, outh, Lq, Len, total);

    gemm_wmma_kernel<<<dim3(4, rt8(qrows)), 256, 0, stream>>>(
        outh, wtw, bw, (float*)d_out, qrows, 256);
}
